// Graph2Vec_54889682043380
// MI455X (gfx1250) — compile-verified, matrix operand order checked by reading, not run on hardware
//
#include <hip/hip_runtime.h>
#include <hip/hip_bf16.h>
#include <stdint.h>

typedef __attribute__((ext_vector_type(16))) _Float16 v16h;
typedef __attribute__((ext_vector_type(8)))  float    v8f;

#define IN_C   128
#define HID_C  128
#define OUT_C  64
#define NGRAPH 128

// Native fp32 atomic add (global_atomic_add_f32, no return) — workspace/output
// are coarse-grained hipMalloc memory, so the "unsafe" path is valid.
__device__ __forceinline__ void atomic_add_f32(float* p, float v) {
    unsafeAtomicAdd(p, v);
}

// ---------------- utility kernels ----------------
__global__ void k_fill(float* p, float v, int n) {
    int i = blockIdx.x * blockDim.x + threadIdx.x;
    if (i < n) p[i] = v;
}

__global__ void k_edge_deg(const int* __restrict__ dst, float* deg, int E) {
    int e = blockIdx.x * blockDim.x + threadIdx.x;
    if (e < E) atomic_add_f32(&deg[dst[e]], 1.0f);
}

__global__ void k_rsqrt(const float* __restrict__ deg, float* dinv, int n) {
    int i = blockIdx.x * blockDim.x + threadIdx.x;
    if (i < n) dinv[i] = rsqrtf(deg[i]);   // deg >= 1 (self-loop)
}

// Pack row-major f32 W[K][N] into WMMA B-fragment order (f16).
// Fragment f = kt*(N/16)+nt occupies 512 halves: lane-major, 16 contiguous per lane.
// Per-lane K map mirrors the A-matrix 16x32 layout from the ISA tables.
__global__ void k_pack_w(const float* __restrict__ W, _Float16* __restrict__ Wh,
                         int K, int N) {
    int idx = blockIdx.x * blockDim.x + threadIdx.x;
    if (idx >= K * N) return;
    int frag   = idx >> 9;        // /512
    int within = idx & 511;
    int lane   = within >> 4;
    int i      = within & 15;
    int ntiles = N >> 4;
    int kt = frag / ntiles, nt = frag % ntiles;
    int n  = (nt << 4) + (lane & 15);
    int kk = (i < 8 ? i : i + 8) + ((lane >> 4) << 3);
    int k  = (kt << 5) + kk;
    Wh[idx] = (_Float16)W[k * N + n];
}

// ---------------- WMMA GEMM: C[M,N] = op(A)[M,K] * B[K,N] ----------------
// One wave per 16-row strip; NT accumulators cover N = NT*16.
template <int NT, bool RELU>
__global__ void k_gemm(const float* __restrict__ A, const _Float16* __restrict__ Bh,
                       float* __restrict__ C, int M, int K, int N) {
    int wave = (int)((blockIdx.x * blockDim.x + threadIdx.x) >> 5);
    int lane = threadIdx.x & 31;
    int mtiles = M >> 4;
    if (wave >= mtiles) return;

    int m0  = wave << 4;
    int hi  = lane >> 4;           // half-wave select
    int row = m0 + (lane & 15);

    v8f acc[NT];
    v8f zero = {0.f,0.f,0.f,0.f,0.f,0.f,0.f,0.f};
    #pragma unroll
    for (int j = 0; j < NT; ++j) acc[j] = zero;

    const v16h* bp = reinterpret_cast<const v16h*>(Bh);
    int ktiles = K >> 5;
    for (int kt = 0; kt < ktiles; ++kt) {
        int kbase = (kt << 5) + (hi << 3);
        const float4* p = reinterpret_cast<const float4*>(A + (long)row * K + kbase);
        float4 x0 = p[0], x1 = p[1];   // K = kbase .. kbase+7
        float4 x2 = p[4], x3 = p[5];   // K = kbase+16 .. kbase+23
        if (RELU) {
            x0.x=fmaxf(x0.x,0.f); x0.y=fmaxf(x0.y,0.f); x0.z=fmaxf(x0.z,0.f); x0.w=fmaxf(x0.w,0.f);
            x1.x=fmaxf(x1.x,0.f); x1.y=fmaxf(x1.y,0.f); x1.z=fmaxf(x1.z,0.f); x1.w=fmaxf(x1.w,0.f);
            x2.x=fmaxf(x2.x,0.f); x2.y=fmaxf(x2.y,0.f); x2.z=fmaxf(x2.z,0.f); x2.w=fmaxf(x2.w,0.f);
            x3.x=fmaxf(x3.x,0.f); x3.y=fmaxf(x3.y,0.f); x3.z=fmaxf(x3.z,0.f); x3.w=fmaxf(x3.w,0.f);
        }
        v16h a;
        a[0]=(_Float16)x0.x; a[1]=(_Float16)x0.y; a[2]=(_Float16)x0.z; a[3]=(_Float16)x0.w;
        a[4]=(_Float16)x1.x; a[5]=(_Float16)x1.y; a[6]=(_Float16)x1.z; a[7]=(_Float16)x1.w;
        a[8]=(_Float16)x2.x; a[9]=(_Float16)x2.y; a[10]=(_Float16)x2.z; a[11]=(_Float16)x2.w;
        a[12]=(_Float16)x3.x; a[13]=(_Float16)x3.y; a[14]=(_Float16)x3.z; a[15]=(_Float16)x3.w;

        #pragma unroll
        for (int nt = 0; nt < NT; ++nt) {
            v16h b = bp[(size_t)(kt * NT + nt) * 32 + lane];
            acc[nt] = __builtin_amdgcn_wmma_f32_16x16x32_f16(
                false, a, false, b, (short)0, acc[nt], false, false);
        }
    }

    int mrow = m0 + (hi << 3);
    #pragma unroll
    for (int nt = 0; nt < NT; ++nt) {
        int n = (nt << 4) + (lane & 15);
        #pragma unroll
        for (int j = 0; j < 8; ++j)
            C[(long)(mrow + j) * N + n] = acc[nt][j];
    }
}

// ---------------- GCN propagate ----------------
// acc[i][c] = H[i][c]*dinv[i]^2 + bias[c]   (self-loop message + bias)
__global__ void k_self_bias(const float* __restrict__ H, const float* __restrict__ dinv,
                            const float* __restrict__ bias, float* __restrict__ acc,
                            int Nn, int C) {
    int idx = blockIdx.x * blockDim.x + threadIdx.x;
    if (idx >= Nn * C) return;
    int i = idx / C, c = idx % C;
    float d = dinv[i];
    acc[idx] = H[idx] * d * d + bias[c];
}

// one wave per edge; lane owns CPE/32 consecutive channels (vector gather + native atomics)
template <int CPE>
__global__ void k_edge_prop(const int* __restrict__ src, const int* __restrict__ dst,
                            const float* __restrict__ H, const float* __restrict__ dinv,
                            float* acc, int E) {
    int wave = (int)((blockIdx.x * blockDim.x + threadIdx.x) >> 5);
    int lane = threadIdx.x & 31;
    if (wave >= E) return;
    int s = src[wave], d = dst[wave];
    float norm = dinv[s] * dinv[d];
    constexpr int FPL = CPE / 32;
    const float* hs = H   + (long)s * CPE + lane * FPL;
    float*       ad = acc + (long)d * CPE + lane * FPL;
    if constexpr (FPL == 4) {
        float4 hv = *reinterpret_cast<const float4*>(hs);   // global_load_b128
        atomic_add_f32(&ad[0], hv.x * norm);
        atomic_add_f32(&ad[1], hv.y * norm);
        atomic_add_f32(&ad[2], hv.z * norm);
        atomic_add_f32(&ad[3], hv.w * norm);
    } else {
        float2 hv = *reinterpret_cast<const float2*>(hs);   // global_load_b64
        atomic_add_f32(&ad[0], hv.x * norm);
        atomic_add_f32(&ad[1], hv.y * norm);
    }
}

// ---------------- global mean pool ----------------
__global__ void k_pool_add(const int* __restrict__ batch, const float* __restrict__ A,
                           float* pool, float* counts, int Nn) {
    int idx = blockIdx.x * blockDim.x + threadIdx.x;
    if (idx >= Nn * OUT_C) return;
    int i = idx >> 6, c = idx & 63;
    int g = batch[i];
    atomic_add_f32(&pool[g * OUT_C + c], A[idx]);
    if (c == 0) atomic_add_f32(&counts[g], 1.0f);
}

__global__ void k_pool_final(const float* __restrict__ pool, const float* __restrict__ counts,
                             float* __restrict__ out) {
    int idx = blockIdx.x * blockDim.x + threadIdx.x;
    if (idx >= NGRAPH * OUT_C) return;
    out[idx] = pool[idx] / fmaxf(counts[idx >> 6], 1.0f);
}

// ---------------- host launch ----------------
extern "C" void kernel_launch(void* const* d_in, const int* in_sizes, int n_in,
                              void* d_out, int out_size, void* d_ws, size_t ws_size,
                              hipStream_t stream) {
    const float* x   = (const float*)d_in[0];
    const float* W1  = (const float*)d_in[1];
    const float* b1  = (const float*)d_in[2];
    const float* W2  = (const float*)d_in[3];
    const float* b2  = (const float*)d_in[4];
    const int*   ei  = (const int*)d_in[5];
    const int*   bat = (const int*)d_in[6];

    const int Nn = in_sizes[0] / IN_C;   // 50000
    const int E  = in_sizes[5] / 2;      // 625000
    const int* src = ei;
    const int* dst = ei + E;

    char* w = (char*)d_ws;
    size_t off = 0;
    auto alloc = [&](size_t bytes) -> void* {
        void* p = w + off;
        off = (off + bytes + 255) & ~(size_t)255;
        return p;
    };
    float*    deg   = (float*)alloc((size_t)Nn * 4);
    float*    dinv  = (float*)alloc((size_t)Nn * 4);
    _Float16* W1h   = (_Float16*)alloc((size_t)IN_C * HID_C * 2);
    _Float16* W2h   = (_Float16*)alloc((size_t)HID_C * OUT_C * 2);
    float*    h1    = (float*)alloc((size_t)Nn * HID_C * 4);  // also reused as h2
    float*    a1    = (float*)alloc((size_t)Nn * HID_C * 4);  // also reused as a2
    float*    pool  = (float*)alloc((size_t)NGRAPH * OUT_C * 4);
    float*    cnts  = (float*)alloc((size_t)NGRAPH * 4);
    float*    h2 = h1;   // h1 dead after propagate1
    float*    a2 = a1;   // a1 dead after gemm2

    const int T = 256;
    // degrees (with self-loop = 1)
    k_fill<<<(Nn + T - 1) / T, T, 0, stream>>>(deg, 1.0f, Nn);
    k_edge_deg<<<(E + T - 1) / T, T, 0, stream>>>(dst, deg, E);
    k_rsqrt<<<(Nn + T - 1) / T, T, 0, stream>>>(deg, dinv, Nn);

    // pack weights to f16 B-fragments
    k_pack_w<<<(IN_C * HID_C + T - 1) / T, T, 0, stream>>>(W1, W1h, IN_C, HID_C);
    k_pack_w<<<(HID_C * OUT_C + T - 1) / T, T, 0, stream>>>(W2, W2h, HID_C, OUT_C);

    // layer 1: h1 = X @ W1   (WMMA)
    {
        int waves = Nn / 16;
        int blocks = (waves * 32 + T - 1) / T;
        k_gemm<HID_C / 16, false><<<blocks, T, 0, stream>>>(x, W1h, h1, Nn, IN_C, HID_C);
    }
    // propagate1: a1 = scatter(norm * h1[src]) + selfloop + b1
    k_self_bias<<<(Nn * HID_C + T - 1) / T, T, 0, stream>>>(h1, dinv, b1, a1, Nn, HID_C);
    k_edge_prop<HID_C><<<(E + 7) / 8, T, 0, stream>>>(src, dst, h1, dinv, a1, E);

    // layer 2: h2 = relu(a1) @ W2   (WMMA, relu fused into A load)
    {
        int waves = Nn / 16;
        int blocks = (waves * 32 + T - 1) / T;
        k_gemm<OUT_C / 16, true><<<blocks, T, 0, stream>>>(a1, W2h, h2, Nn, HID_C, OUT_C);
    }
    // propagate2
    k_self_bias<<<(Nn * OUT_C + T - 1) / T, T, 0, stream>>>(h2, dinv, b2, a2, Nn, OUT_C);
    k_edge_prop<OUT_C><<<(E + 7) / 8, T, 0, stream>>>(src, dst, h2, dinv, a2, E);

    // global mean pool
    k_fill<<<(NGRAPH * OUT_C + T - 1) / T, T, 0, stream>>>(pool, 0.0f, NGRAPH * OUT_C);
    k_fill<<<1, T, 0, stream>>>(cnts, 0.0f, NGRAPH);
    k_pool_add<<<(Nn * OUT_C + T - 1) / T, T, 0, stream>>>(bat, a2, pool, cnts, Nn);
    k_pool_final<<<(NGRAPH * OUT_C + T - 1) / T, T, 0, stream>>>(pool, cnts, (float*)d_out);
}
